// AttentionBlockWithWeights_11029476016814
// MI455X (gfx1250) — matrix operand order repaired and checked by reading, not compiled, not run on hardware
//
#include <hip/hip_runtime.h>

typedef __attribute__((ext_vector_type(16))) _Float16 v16h;
typedef __attribute__((ext_vector_type(8)))  _Float16 v8h;
typedef __attribute__((ext_vector_type(8)))  float    v8f;

#define NB   32
#define CCH  64
#define HWP  1024
#define CHW  65536      /* C*H*W  */
#define HW2  1048576    /* HW*HW  */
#define EPSV 1e-5f

// ---------------------------------------------------------------------------
// Kernel 1: LayerNorm statistics. One block per (tensor t, sample n).
// ---------------------------------------------------------------------------
__global__ void __launch_bounds__(256)
ln_stats_kernel(const float* __restrict__ q, const float* __restrict__ k,
                const float* __restrict__ v, float* __restrict__ stats) {
  int bx = blockIdx.x;               // t*32 + n
  int t = bx >> 5, n = bx & 31;
  const float* x = (t == 0) ? q : (t == 1) ? k : v;
  x += (size_t)n * CHW;
  float s = 0.f, ss = 0.f;
  for (int i = threadIdx.x; i < CHW; i += 256) {
    float xv = x[i]; s += xv; ss += xv * xv;
  }
  __shared__ float rs[256], rss[256];
  rs[threadIdx.x] = s; rss[threadIdx.x] = ss;
  __syncthreads();
  for (int off = 128; off > 0; off >>= 1) {
    if (threadIdx.x < off) {
      rs[threadIdx.x]  += rs[threadIdx.x + off];
      rss[threadIdx.x] += rss[threadIdx.x + off];
    }
    __syncthreads();
  }
  if (threadIdx.x == 0) {
    float mu  = rs[0]  * (1.f / CHW);
    float var = rss[0] * (1.f / CHW) - mu * mu;
    stats[(t * NB + n) * 2 + 0] = mu;
    stats[(t * NB + n) * 2 + 1] = rsqrtf(var + EPSV);
  }
}

// ---------------------------------------------------------------------------
// Kernel 2: normalize + 1x1-conv projection.
//   - qn (normalized query) -> f32 natural layout (residual path)
//   - Q  -> f16 in WMMA *B*-fragment-swizzled global layout
//   - K  -> f16 in WMMA *A*-fragment-swizzled global layout
//   - V  -> f16 natural [c][p] layout (A-frags of GEMM2 are row-contiguous)
// Fragment layout: [n][tile(64)][kchunk(2)][lane(32)][e(16)] halves.
//   B-swizzle: elem(c',col): lane = col + ((c'>>3)&1)*16, e = (c'&7) | ((c'>>4)<<3)
//   A-swizzle: elem(c',m)  : lane = m + ((c'>>3)&1)*16, e = (c'&7) | ((c'>>4)<<3)
// ---------------------------------------------------------------------------
__global__ void __launch_bounds__(256)
norm_proj_kernel(const float* __restrict__ xq, const float* __restrict__ xk,
                 const float* __restrict__ xv,
                 const float* __restrict__ ln1w, const float* __restrict__ ln1b,
                 const float* __restrict__ ln2w, const float* __restrict__ ln2b,
                 const float* __restrict__ ln3w, const float* __restrict__ ln3b,
                 const float* __restrict__ wq, const float* __restrict__ bq,
                 const float* __restrict__ wk, const float* __restrict__ bk,
                 const float* __restrict__ wv, const float* __restrict__ bv,
                 const float* __restrict__ stats,
                 float* __restrict__ qn,
                 _Float16* __restrict__ qf, _Float16* __restrict__ kf,
                 _Float16* __restrict__ vf) {
  __shared__ float xs[64][128];
  __shared__ float Wl[64][64];
  __shared__ float bl[64];
  int bx = blockIdx.x;
  int n = bx >> 3, pt = bx & 7;
  int tid = threadIdx.x;
  int tx = tid & 127, ty = tid >> 7;
  const float* xin[3] = {xq, xk, xv};
  const float* lw[3]  = {ln1w, ln2w, ln3w};
  const float* lb[3]  = {ln1b, ln2b, ln3b};
  const float* Wm[3]  = {wq, wk, wv};
  const float* Bs[3]  = {bq, bk, bv};
#pragma unroll
  for (int t = 0; t < 3; ++t) {
    float mu   = stats[(t * NB + n) * 2 + 0];
    float rstd = stats[(t * NB + n) * 2 + 1];
    for (int r = 0; r < 32; ++r) {
      int lin = tid + 256 * r;             // 0..8191
      int c = lin >> 7, p = lin & 127;
      int pg = pt * 128 + p;
      float xv = xin[t][(size_t)n * CHW + c * HWP + pg];
      float xn = (xv - mu) * rstd * lw[t][c * HWP + pg] + lb[t][c * HWP + pg];
      xs[c][p] = xn;
      if (t == 0) qn[(size_t)n * CHW + c * HWP + pg] = xn;
    }
    for (int r = 0; r < 16; ++r) {
      int lin = tid + 256 * r;             // 0..4095
      Wl[lin >> 6][lin & 63] = Wm[t][lin];
    }
    if (tid < 64) bl[tid] = Bs[t][tid];
    __syncthreads();
    float acc[32];
    // output channel o = ty*32 + oo  ->  kchunk kc = ty, c' = oo
    int ob = ty * 32;
#pragma unroll
    for (int oo = 0; oo < 32; ++oo) acc[oo] = bl[ob + oo];
    for (int c = 0; c < 64; ++c) {
      float xv = xs[c][tx];
#pragma unroll
      for (int oo = 0; oo < 32; ++oo) acc[oo] += Wl[ob + oo][c] * xv;
    }
    int pg = pt * 128 + tx;
    int kc = ty;
    if (t == 0) {
      // Q: B-fragment swizzle. Four contiguous 8-half runs.
      int jt = pg >> 4, col = pg & 15;
      size_t base = ((size_t)(n * 64 + jt) * 2 + kc) * 512;
      v8h pk;
#pragma unroll
      for (int i = 0; i < 8; ++i) pk[i] = (_Float16)acc[i];        // c'=0..7  -> lane=col,    e=0..7
      *reinterpret_cast<v8h*>(qf + base + col * 16 + 0) = pk;
#pragma unroll
      for (int i = 0; i < 8; ++i) pk[i] = (_Float16)acc[8 + i];    // c'=8..15 -> lane=col+16, e=0..7
      *reinterpret_cast<v8h*>(qf + base + (col + 16) * 16 + 0) = pk;
#pragma unroll
      for (int i = 0; i < 8; ++i) pk[i] = (_Float16)acc[16 + i];   // c'=16..23-> lane=col,    e=8..15
      *reinterpret_cast<v8h*>(qf + base + col * 16 + 8) = pk;
#pragma unroll
      for (int i = 0; i < 8; ++i) pk[i] = (_Float16)acc[24 + i];   // c'=24..31-> lane=col+16, e=8..15
      *reinterpret_cast<v8h*>(qf + base + (col + 16) * 16 + 8) = pk;
    } else if (t == 1) {
      // K: A-fragment swizzle. Four contiguous 8-half runs.
      int it = pg >> 4, m = pg & 15;
      size_t base = ((size_t)(n * 64 + it) * 2 + kc) * 512;
      v8h pk;
#pragma unroll
      for (int i = 0; i < 8; ++i) pk[i] = (_Float16)acc[i];        // c'=0..7  -> lane=m,    e=0..7
      *reinterpret_cast<v8h*>(kf + base + m * 16 + 0) = pk;
#pragma unroll
      for (int i = 0; i < 8; ++i) pk[i] = (_Float16)acc[8 + i];    // c'=8..15 -> lane=m+16, e=0..7
      *reinterpret_cast<v8h*>(kf + base + (m + 16) * 16 + 0) = pk;
#pragma unroll
      for (int i = 0; i < 8; ++i) pk[i] = (_Float16)acc[16 + i];   // c'=16..23-> lane=m,    e=8..15
      *reinterpret_cast<v8h*>(kf + base + m * 16 + 8) = pk;
#pragma unroll
      for (int i = 0; i < 8; ++i) pk[i] = (_Float16)acc[24 + i];   // c'=24..31-> lane=m+16, e=8..15
      *reinterpret_cast<v8h*>(kf + base + (m + 16) * 16 + 8) = pk;
    } else {
      // V: natural [c][p]
#pragma unroll
      for (int oo = 0; oo < 32; ++oo)
        vf[(size_t)n * CHW + (ob + oo) * HWP + pg] = (_Float16)acc[oo];
    }
    __syncthreads();
  }
}

// ---------------------------------------------------------------------------
// Kernel 3: S = (1/sqrt(C)) * K^T Q via WMMA. One wave per 16(i) x 64(j)
// strip; fragments loaded directly from swizzled global with b128 loads.
// No LDS, no barriers.
// ---------------------------------------------------------------------------
__global__ void __launch_bounds__(256)
attn_logits_kernel(const _Float16* __restrict__ kfrag,
                   const _Float16* __restrict__ qfrag,
                   float* __restrict__ attn) {
  int gid  = blockIdx.x * 8 + (threadIdx.x >> 5);  // wave id, 0..32767
  int lane = threadIdx.x & 31, l16 = lane & 15;
  bool hi = lane >= 16;
  int n = gid >> 10; int rem = gid & 1023;
  int it = rem >> 4;                 // 16-row i tile
  int j0 = (rem & 15) * 64;          // 64-column j strip
  int i0 = it * 16;
  v8f acc[4];
#pragma unroll
  for (int jb = 0; jb < 4; ++jb)
#pragma unroll
    for (int e = 0; e < 8; ++e) acc[jb][e] = 0.f;

#pragma unroll
  for (int kc = 0; kc < 2; ++kc) {
    union { v16h v; v8h h[2]; } a;
    const _Float16* pa = kfrag + ((size_t)(n * 64 + it) * 2 + kc) * 512 + lane * 16;
    a.h[0] = *reinterpret_cast<const v8h*>(pa);
    a.h[1] = *reinterpret_cast<const v8h*>(pa + 8);
#pragma unroll
    for (int jb = 0; jb < 4; ++jb) {
      union { v16h v; v8h h[2]; } b;
      const _Float16* pb =
          qfrag + ((size_t)(n * 64 + (j0 >> 4) + jb) * 2 + kc) * 512 + lane * 16;
      b.h[0] = *reinterpret_cast<const v8h*>(pb);
      b.h[1] = *reinterpret_cast<const v8h*>(pb + 8);
      acc[jb] = __builtin_amdgcn_wmma_f32_16x16x32_f16(
          false, a.v, false, b.v, (short)0, acc[jb], false, false);
    }
  }
  const float scale = 0.125f;        // 1/sqrt(64)
#pragma unroll
  for (int jb = 0; jb < 4; ++jb)
#pragma unroll
    for (int r = 0; r < 8; ++r) {
      int i = i0 + (hi ? r + 8 : r);
      int j = j0 + jb * 16 + l16;
      attn[(size_t)n * HW2 + (size_t)i * HWP + j] = acc[jb][r] * scale;
    }
}

// ---------------------------------------------------------------------------
// Kernel 4: online softmax statistics per column (axis i). Block = (n, 64 j's).
// ---------------------------------------------------------------------------
__global__ void __launch_bounds__(256)
softmax_stats_kernel(const float* __restrict__ attn,
                     float* __restrict__ cmax, float* __restrict__ cinv) {
  int bx = blockIdx.x; int n = bx >> 4, j0 = (bx & 15) * 64;
  int tx = threadIdx.x & 63, ty = threadIdx.x >> 6;
  float m = -3.0e38f, l = 0.f;
  for (int i = ty; i < HWP; i += 4) {
    float s = attn[(size_t)n * HW2 + (size_t)i * HWP + j0 + tx];
    float mn = fmaxf(m, s);
    l = l * __expf(m - mn) + __expf(s - mn);
    m = mn;
  }
  __shared__ float sm[4][64], sl[4][64];
  sm[ty][tx] = m; sl[ty][tx] = l;
  __syncthreads();
  if (ty == 0) {
    float M = sm[0][tx], L = sl[0][tx];
#pragma unroll
    for (int t = 1; t < 4; ++t) {
      float mk = sm[t][tx];
      float Mn = fmaxf(M, mk);
      L = L * __expf(M - Mn) + sl[t][tx] * __expf(mk - Mn);
      M = Mn;
    }
    cmax[n * HWP + j0 + tx] = M;
    cinv[n * HWP + j0 + tx] = 1.0f / L;
  }
}

// ---------------------------------------------------------------------------
// Kernel 5: normalize attn in place + out = V * P (WMMA), x = qn + out.
// One wave per (n, 16-column tile). Each lane computes exactly the P values
// of its own B fragment (exp in registers, write-back in place) — no LDS,
// no cross-lane traffic. V A-fragments load contiguously from [c][i] f16.
// ---------------------------------------------------------------------------
__global__ void __launch_bounds__(256)
attn_out_kernel(float* __restrict__ attn,
                const _Float16* __restrict__ vf,
                const float* __restrict__ qn,
                const float* __restrict__ cmax, const float* __restrict__ cinv,
                float* __restrict__ xout) {
  int gid  = blockIdx.x * 8 + (threadIdx.x >> 5);  // wave id, 0..2047
  int lane = threadIdx.x & 31, l16 = lane & 15;
  bool hi = lane >= 16;
  int n = gid >> 6;
  int j0 = (gid & 63) * 16;
  int jc = j0 + l16;                 // this lane's column
  float cmv = cmax[n * HWP + jc];
  float civ = cinv[n * HWP + jc];
  v8f acc[4];
#pragma unroll
  for (int mt = 0; mt < 4; ++mt)
#pragma unroll
    for (int e = 0; e < 8; ++e) acc[mt][e] = 0.f;

  for (int ic = 0; ic < 32; ++ic) {
    int i0 = ic * 32;
    // Build the normalized-P B fragment in registers; write attn in place.
    v16h b;
#pragma unroll
    for (int e = 0; e < 16; ++e) {
      int irow = i0 + (hi ? 8 : 0) + e + (e >= 8 ? 8 : 0);   // K=8h+e (e<8), 16+8h+(e-8) (e>=8)
      size_t idx = (size_t)n * HW2 + (size_t)irow * HWP + jc;
      float p = __expf(attn[idx] - cmv) * civ;
      attn[idx] = p;
      b[e] = (_Float16)p;
    }
#pragma unroll
    for (int mt = 0; mt < 4; ++mt) {
      union { v16h v; v8h h[2]; } a;
      const _Float16* pa = vf + (size_t)n * CHW +
                           (size_t)(mt * 16 + l16) * HWP + i0 + (hi ? 8 : 0);
      a.h[0] = *reinterpret_cast<const v8h*>(pa);
      a.h[1] = *reinterpret_cast<const v8h*>(pa + 16);
      acc[mt] = __builtin_amdgcn_wmma_f32_16x16x32_f16(
          false, a.v, false, b, (short)0, acc[mt], false, false);
    }
  }
#pragma unroll
  for (int mt = 0; mt < 4; ++mt)
#pragma unroll
    for (int r = 0; r < 8; ++r) {
      int c = mt * 16 + (hi ? r + 8 : r);
      size_t idx = (size_t)n * CHW + (size_t)c * HWP + jc;
      xout[idx] = qn[idx] + acc[mt][r];
    }
}

// ---------------------------------------------------------------------------
extern "C" void kernel_launch(void* const* d_in, const int* in_sizes, int n_in,
                              void* d_out, int out_size, void* d_ws, size_t ws_size,
                              hipStream_t stream) {
  const float* query = (const float*)d_in[0];
  const float* key   = (const float*)d_in[1];
  const float* value = (const float*)d_in[2];
  const float* ln1w  = (const float*)d_in[3];
  const float* ln1b  = (const float*)d_in[4];
  const float* ln2w  = (const float*)d_in[5];
  const float* ln2b  = (const float*)d_in[6];
  const float* ln3w  = (const float*)d_in[7];
  const float* ln3b  = (const float*)d_in[8];
  const float* wq    = (const float*)d_in[9];
  const float* bq    = (const float*)d_in[10];
  const float* wk    = (const float*)d_in[11];
  const float* bk    = (const float*)d_in[12];
  const float* wv    = (const float*)d_in[13];
  const float* bv    = (const float*)d_in[14];

  float* xout = (float*)d_out;
  float* attn = xout + (size_t)NB * CHW;     // second output region (128 MB)

  char* ws = (char*)d_ws;
  float*    qn    = (float*)(ws);                    // 8 MB  f32 normalized query
  _Float16* qf    = (_Float16*)(ws + 8388608);       // 4 MB  f16 Q (B-swizzled)
  _Float16* kf    = (_Float16*)(ws + 12582912);      // 4 MB  f16 K (A-swizzled)
  _Float16* vfp   = (_Float16*)(ws + 16777216);      // 4 MB  f16 V (natural)
  float*    stats = (float*)(ws + 20971520);         // 768 B (mu, rstd) x3 x32
  float*    cmaxp = (float*)(ws + 20972544);         // 128 KB column max
  float*    cinvp = (float*)(ws + 21103616);         // 128 KB column 1/sum

  ln_stats_kernel<<<dim3(96), dim3(256), 0, stream>>>(query, key, value, stats);

  norm_proj_kernel<<<dim3(256), dim3(256), 0, stream>>>(
      query, key, value, ln1w, ln1b, ln2w, ln2b, ln3w, ln3b,
      wq, bq, wk, bk, wv, bv, stats, qn, qf, kf, vfp);

  attn_logits_kernel<<<dim3(4096), dim3(256), 0, stream>>>(kf, qf, attn);

  softmax_stats_kernel<<<dim3(512), dim3(256), 0, stream>>>(attn, cmaxp, cinvp);

  attn_out_kernel<<<dim3(256), dim3(256), 0, stream>>>(attn, vfp, qn, cmaxp, cinvp, xout);
}
